// _ConvExpert_82094004896560
// MI455X (gfx1250) — compile-verified, matrix operand order checked by reading, not run on hardware
//
#include <hip/hip_runtime.h>

// ---------------------------------------------------------------------------
// MoE dilated conv1d (E=8, DM=512, DH=1024, K=3, dil=2, N=128, L=128)
// bf16x3 split GEMM on v_wmma_f32_16x16x32_bf16 with an ASYNC double-buffered
// global->LDS pipeline (GLOBAL_LOAD_ASYNC_TO_LDS_B128 + s_wait_asynccnt):
//   block (256 thr, 8 waves) = 128(M) x 64(L) macro-tile,
//   per K-step: issue async stage(s+1) into buffer^1, compute stage(s).
// ---------------------------------------------------------------------------

typedef __attribute__((ext_vector_type(16))) __bf16 v16bf;
typedef __attribute__((ext_vector_type(8)))  __bf16 v8bf;
typedef __attribute__((ext_vector_type(8)))  float  v8f;

#define E_    8
#define DM_   512
#define DH_   1024
#define L_    128
#define N_    128
#define KTAPS 3
#define DIL_  2

#define APITCH 40                 // 32 K elems + 8 pad (80B pitch) for LDS banks
#define BPITCH 40
#define ABUF   (128 * APITCH)     // elements per A panel (one of hi/lo)
#define BBUF   (64 * BPITCH)      // elements per B panel
#define BUFSZ  (2 * ABUF + 2 * BBUF)  // one pipeline buffer (Ah,Al,Bh,Bl)

static __device__ __forceinline__ float gelu_tanh(float x) {
    // jax.nn.gelu default (approximate=True)
    const float c0 = 0.7978845608028654f;
    float x3 = x * x * x;
    return 0.5f * x * (1.0f + tanhf(c0 * (x + 0.044715f * x3)));
}

// Async DMA: 16B per lane, global -> LDS. LDS address = low 32 bits of the
// flat shared-memory pointer. ASYNCcnt-tracked.
static __device__ __forceinline__ void async_copy_b128(const void* g, void* l) {
    unsigned lds = (unsigned)(uintptr_t)l;
    unsigned long long ga = (unsigned long long)(uintptr_t)g;
    asm volatile("global_load_async_to_lds_b128 %0, %1, off"
                 :: "v"(lds), "v"(ga) : "memory");
}

static __device__ __forceinline__ void wait_async0() {
#if __has_builtin(__builtin_amdgcn_s_wait_asynccnt)
    __builtin_amdgcn_s_wait_asynccnt(0);
#else
    asm volatile("s_wait_asynccnt 0x0" ::: "memory");
#endif
}

// Fragment layout (wave32 bf16 WMMA):
//   lanes 0-15 : row/col = lane,    K = kh+0..7 (elems 0-7), kh+16..23 (elems 8-15), kh=0
//   lanes 16-31: row/col = lane-16, same pattern with kh=8
static __device__ __forceinline__ v16bf lds_frag(const __bf16* p) {
    v8bf a = *(const v8bf*)(p);        // K = kh + 0..7
    v8bf b = *(const v8bf*)(p + 16);   // K = kh + 16..23
    v16bf r;
#pragma unroll
    for (int i = 0; i < 8; ++i) { r[i] = a[i]; r[i + 8] = b[i]; }
    return r;
}

// ---- prep: split fp32 weights [E][COUT][CIN][3] -> bf16 hi/lo [E][3][COUT][CIN]
__global__ __launch_bounds__(256) void split_w(const float* __restrict__ W,
                                               __bf16* __restrict__ WH,
                                               __bf16* __restrict__ WL,
                                               int COUT, int CIN) {
    size_t idx = (size_t)blockIdx.x * blockDim.x + threadIdx.x;
    size_t total = (size_t)E_ * COUT * CIN * KTAPS;
    if (idx >= total) return;
    int t = (int)(idx % KTAPS); size_t r = idx / KTAPS;
    int c = (int)(r % CIN);     r /= CIN;
    int o = (int)(r % COUT);
    int e = (int)(r / COUT);
    float v = W[idx];
    __bf16 h = (__bf16)v;
    size_t d = (((size_t)e * KTAPS + t) * COUT + o) * CIN + c;
    WH[d] = h;
    WL[d] = (__bf16)(v - (float)h);
}

// ---- prep: split+transpose fp32 x [N][C][L] -> bf16 hi/lo [N][L][C]
__global__ __launch_bounds__(256) void split_x(const float* __restrict__ X,
                                               __bf16* __restrict__ XH,
                                               __bf16* __restrict__ XL, int C) {
    size_t idx = (size_t)blockIdx.x * blockDim.x + threadIdx.x;
    size_t total = (size_t)N_ * C * L_;
    if (idx >= total) return;
    int l = (int)(idx % L_); size_t r = idx / L_;
    int c = (int)(r % C);
    int n = (int)(r / C);
    float v = X[idx];
    __bf16 h = (__bf16)v;
    size_t d = ((size_t)n * L_ + l) * C + c;
    XH[d] = h;
    XL[d] = (__bf16)(v - (float)h);
}

// ---------------------------------------------------------------------------
// Block computes 128(M=o) x 64(N=l) for one sample; wave = 32x32 sub-tile.
// B (activations) split-transposed [N][L][CIN]; A (weights) [E][3][COUT][CIN].
// SPLIT_OUT: write bf16 hi/lo transposed [N][L][COUT]; else fp32 [N][COUT][L].
// ---------------------------------------------------------------------------
template <int COUT, int CIN, bool GELU, bool SPLIT_OUT>
__global__ __launch_bounds__(256) void moe_conv_wmma(
    const __bf16* __restrict__ BH, const __bf16* __restrict__ BL,
    const __bf16* __restrict__ WH, const __bf16* __restrict__ WL,
    const float* __restrict__ Bias, const int* __restrict__ counts,
    __bf16* __restrict__ YH, __bf16* __restrict__ YL,
    float* __restrict__ Yf) {
    __shared__ __bf16 smem[2 * BUFSZ];   // double-buffered panels, ~60 KB

    const int n = blockIdx.x;

    // expert id = searchsorted(cumsum(counts), n, 'right')  (block-uniform)
    int eid = 0, csum = 0;
#pragma unroll
    for (int e = 0; e < E_; ++e) { csum += counts[e]; if (n >= csum) eid = e + 1; }

    const int tid  = threadIdx.x;
    const int lane = tid & 31;
    const int wave = tid >> 5;           // 8 waves (wave32)
    const int mw   = wave >> 1;          // 0..3  (M sub-tile)
    const int lw   = wave & 1;           // 0..1  (L sub-tile)
    const int mb   = (blockIdx.y >> 1) * 128;   // macro-tile M origin
    const int lb   = (blockIdx.y & 1) * 64;     // macro-tile L origin

    const int row   = lane & 15;
    const int khalf = (lane >> 4) * 8;

    const __bf16* Bh = BH + (size_t)n * L_ * CIN;
    const __bf16* Bl = BL + (size_t)n * L_ * CIN;
    const __bf16* Wh = WH + (size_t)eid * KTAPS * COUT * CIN;
    const __bf16* Wl = WL + (size_t)eid * KTAPS * COUT * CIN;

    // staging assignments (block-wide, 256 threads)
    const int arow = tid >> 1;            // 0..127
    const int aseg = (tid & 1) * 16;      // 0 or 16 (elements)
    const int bcol = tid >> 2;            // 0..63
    const int bseg = (tid & 3) * 8;       // 0,8,16,24 (elements)

    const int CSTEPS = CIN / 32;
    const int KSTEPS = KTAPS * CSTEPS;

    // issue async global->LDS loads for K-step s into buffer buf
    auto stage = [&](int s, int buf) {
        const int tap = s / CSTEPS;
        const int c0  = (s - tap * CSTEPS) * 32;
        __bf16* dAh = smem + buf * BUFSZ;
        __bf16* dAl = dAh + ABUF;
        __bf16* dBh = dAl + ABUF;
        __bf16* dBl = dBh + BBUF;
        // A panel: 128 rows x 32 K (hi+lo)
        const size_t go = ((size_t)tap * COUT + mb + arow) * CIN + c0 + aseg;
        async_copy_b128(Wh + go,     dAh + arow * APITCH + aseg);
        async_copy_b128(Wh + go + 8, dAh + arow * APITCH + aseg + 8);
        async_copy_b128(Wl + go,     dAl + arow * APITCH + aseg);
        async_copy_b128(Wl + go + 8, dAl + arow * APITCH + aseg + 8);
        __builtin_prefetch(Wh + go + 64, 0, 1);   // next-next line -> global_prefetch
        // B panel: 64 cols x 32 K (hi+lo), dilated shift + 'same' zero pad
        const int shift = (tap - 1) * DIL_;
        const int lsrc  = lb + bcol + shift;
        if (lsrc >= 0 && lsrc < L_) {
            const size_t gb = (size_t)lsrc * CIN + c0 + bseg;
            async_copy_b128(Bh + gb, dBh + bcol * BPITCH + bseg);
            async_copy_b128(Bl + gb, dBl + bcol * BPITCH + bseg);
        } else {
            v8bf z = {};
            *(v8bf*)(dBh + bcol * BPITCH + bseg) = z;
            *(v8bf*)(dBl + bcol * BPITCH + bseg) = z;
        }
    };

    v8f acc[2][2] = {};

    stage(0, 0);
    for (int s = 0; s < KSTEPS; ++s) {
        const int buf = s & 1;
        wait_async0();        // this wave's stage(s) writes retired
        __syncthreads();      // stage(s) visible; compute(s-1) finished everywhere
        if (s + 1 < KSTEPS) stage(s + 1, buf ^ 1);   // overlap with compute(s)

        const __bf16* pAh = smem + buf * BUFSZ;
        const __bf16* pAl = pAh + ABUF;
        const __bf16* pBh = pAl + ABUF;
        const __bf16* pBl = pBh + BBUF;

        v16bf ah[2], al[2], bh[2], bl[2];
#pragma unroll
        for (int mi = 0; mi < 2; ++mi) {
            const int ao = (mw * 32 + mi * 16 + row) * APITCH + khalf;
            ah[mi] = lds_frag(pAh + ao);
            al[mi] = lds_frag(pAl + ao);
        }
#pragma unroll
        for (int li = 0; li < 2; ++li) {
            const int bo = (lw * 32 + li * 16 + row) * BPITCH + khalf;
            bh[li] = lds_frag(pBh + bo);
            bl[li] = lds_frag(pBl + bo);
        }

        // bf16x3 split products into f32 accumulators
#pragma unroll
        for (int mi = 0; mi < 2; ++mi)
#pragma unroll
            for (int li = 0; li < 2; ++li) {
                acc[mi][li] = __builtin_amdgcn_wmma_f32_16x16x32_bf16(
                    false, ah[mi], false, bh[li], (short)0, acc[mi][li], false, false);
                acc[mi][li] = __builtin_amdgcn_wmma_f32_16x16x32_bf16(
                    false, ah[mi], false, bl[li], (short)0, acc[mi][li], false, false);
                acc[mi][li] = __builtin_amdgcn_wmma_f32_16x16x32_bf16(
                    false, al[mi], false, bh[li], (short)0, acc[mi][li], false, false);
            }
    }

    // ---- epilogue: bias (+gelu), store ----
    // C/D layout: VGPR r, lanes 0-15 -> M=r, lanes 16-31 -> M=8+r; N = lane&15
    const float* bp = Bias + (size_t)eid * COUT;
#pragma unroll
    for (int mi = 0; mi < 2; ++mi)
#pragma unroll
        for (int li = 0; li < 2; ++li) {
            const int l  = lb + lw * 32 + li * 16 + (lane & 15);
            const int ob = mb + mw * 32 + mi * 16 + (lane >> 4) * 8;
            if (SPLIT_OUT) {
                v8bf hv, lv;
#pragma unroll
                for (int r = 0; r < 8; ++r) {
                    float v = acc[mi][li][r] + bp[ob + r];
                    if (GELU) v = gelu_tanh(v);
                    __bf16 h = (__bf16)v;
                    hv[r] = h;
                    lv[r] = (__bf16)(v - (float)h);
                }
                const size_t d = ((size_t)n * L_ + l) * COUT + ob;
                *(v8bf*)(YH + d) = hv;
                *(v8bf*)(YL + d) = lv;
            } else {
#pragma unroll
                for (int r = 0; r < 8; ++r) {
                    float v = acc[mi][li][r] + bp[ob + r];
                    if (GELU) v = gelu_tanh(v);
                    Yf[((size_t)n * COUT + ob + r) * L_ + l] = v;
                }
            }
        }
}

extern "C" void kernel_launch(void* const* d_in, const int* in_sizes, int n_in,
                              void* d_out, int out_size, void* d_ws, size_t ws_size,
                              hipStream_t stream) {
    const float* inp = (const float*)d_in[0];  // [N, DM, L]
    const int*   cnt = (const int*)  d_in[1];  // [E]
    const float* w1  = (const float*)d_in[2];  // [E, DH, DM, 3]
    const float* b1  = (const float*)d_in[3];  // [E, DH]
    const float* w2  = (const float*)d_in[4];  // [E, DM, DH, 3]
    const float* b2  = (const float*)d_in[5];  // [E, DM]
    float* out = (float*)d_out;                // [N, DM, L]

    // carve workspace (all regions fully rewritten every call)
    size_t off = 0;
    auto carve = [&](size_t bytes) -> void* {
        void* p = (char*)d_ws + off;
        off += (bytes + 255) & ~(size_t)255;
        return p;
    };
    __bf16* XH  = (__bf16*)carve((size_t)N_ * L_ * DM_ * sizeof(__bf16));
    __bf16* XL  = (__bf16*)carve((size_t)N_ * L_ * DM_ * sizeof(__bf16));
    __bf16* W1H = (__bf16*)carve((size_t)E_ * KTAPS * DH_ * DM_ * sizeof(__bf16));
    __bf16* W1L = (__bf16*)carve((size_t)E_ * KTAPS * DH_ * DM_ * sizeof(__bf16));
    __bf16* W2H = (__bf16*)carve((size_t)E_ * KTAPS * DM_ * DH_ * sizeof(__bf16));
    __bf16* W2L = (__bf16*)carve((size_t)E_ * KTAPS * DM_ * DH_ * sizeof(__bf16));
    __bf16* HH  = (__bf16*)carve((size_t)N_ * L_ * DH_ * sizeof(__bf16));
    __bf16* HL  = (__bf16*)carve((size_t)N_ * L_ * DH_ * sizeof(__bf16));

    // ---- prep ----
    {
        size_t tot = (size_t)N_ * DM_ * L_;
        split_x<<<dim3((unsigned)((tot + 255) / 256)), dim3(256), 0, stream>>>(inp, XH, XL, DM_);
    }
    {
        size_t tot = (size_t)E_ * DH_ * DM_ * KTAPS;
        split_w<<<dim3((unsigned)((tot + 255) / 256)), dim3(256), 0, stream>>>(w1, W1H, W1L, DH_, DM_);
    }
    {
        size_t tot = (size_t)E_ * DM_ * DH_ * KTAPS;
        split_w<<<dim3((unsigned)((tot + 255) / 256)), dim3(256), 0, stream>>>(w2, W2H, W2L, DM_, DH_);
    }

    dim3 blk(256, 1, 1);
    // ---- layer 1: macro-tiles (1024/128) x (128/64) = 8 x 2 per sample ----
    dim3 g1(N_, (DH_ / 128) * (L_ / 64), 1);   // 128 x 16
    moe_conv_wmma<DH_, DM_, true, true><<<g1, blk, 0, stream>>>(
        XH, XL, W1H, W1L, b1, cnt, HH, HL, nullptr);

    // ---- layer 2: macro-tiles (512/128) x 2 = 4 x 2 per sample ----
    dim3 g2(N_, (DM_ / 128) * (L_ / 64), 1);   // 128 x 8
    moe_conv_wmma<DM_, DH_, false, false><<<g2, blk, 0, stream>>>(
        HH, HL, W2H, W2L, b2, cnt, nullptr, nullptr, out);
}
